// AttentionLayer_38903813767774
// MI455X (gfx1250) — matrix operand order repaired and checked by reading, not compiled, not run on hardware
//
#include <hip/hip_runtime.h>
#include <hip/hip_bf16.h>

// GAT layer for MI455X (gfx1250, wave32, WMMA + TDM).
// Softmax is fused into the aggregation GEMM using a fixed per-row shift
// m_i = lrelu(wh1_i + max_j wh2_j)  (valid upper bound since lrelu is
// monotonic), so no online-max rescaling is needed: single pass over adj.
// wh2 (32 KB, broadcast-reused) is staged into LDS via the Tensor Data Mover.

#define GN    8192
#define DIN   512
#define DOUT  256
#define ALPHA 0.2f
#define EPSN  1e-12f

typedef __attribute__((ext_vector_type(16))) __bf16       v16bf;
typedef __attribute__((ext_vector_type(8)))  float        v8f;
typedef __attribute__((ext_vector_type(4)))  unsigned int u32x4;
typedef __attribute__((ext_vector_type(8)))  int          i32x8;
typedef __attribute__((ext_vector_type(4)))  int          i32x4;

__device__ __forceinline__ float lrelu(float x) { return x > 0.f ? x : ALPHA * x; }

__device__ __forceinline__ v8f v8f_zero() {
    v8f z = {0.f, 0.f, 0.f, 0.f, 0.f, 0.f, 0.f, 0.f};
    return z;
}

// ---------------------------------------------------------------------------
// TDM: DMA one contiguous row of `nelem` f32 from global into LDS.
// Builds a D# (groups 0/1) per CDNA5 ISA:
//   g0: count=1 | lds_addr | global_addr | type=2
//   g1: data_size=4B, tensor_dim0=nelem, tensor_dim1=1, tile_dim0=nelem,
//       tile_dim1=1, stride0=nelem
// Tracked with TENSORcnt; caller waits with s_wait_tensorcnt.
// ---------------------------------------------------------------------------
#if __has_builtin(__builtin_amdgcn_tensor_load_to_lds)
#define HAVE_TDM 1
__device__ __forceinline__ void tdm_load_row_f32(unsigned lds_off,
                                                 const float* gsrc, int nelem)
{
    const unsigned long long ga = (unsigned long long)gsrc;
    u32x4 g0;
    g0[0] = 1u;                                            // count=1, user D#
    g0[1] = lds_off;                                       // LDS byte address
    g0[2] = (unsigned)(ga & 0xffffffffu);                  // global_addr[31:0]
    g0[3] = (unsigned)((ga >> 32) & 0x1ffffffu) | (2u << 30); // [56:32] | type=2

    i32x8 g1;
    g1[0] = (int)(2u << 16);                               // data_size = 4B
    g1[1] = (int)(((unsigned)nelem & 0xffffu) << 16);      // tensor_dim0[15:0]
    g1[2] = (int)(((unsigned)nelem >> 16) | (1u << 16));   // dim0[31:16]|dim1=1
    g1[3] = (int)(((unsigned)nelem & 0xffffu) << 16);      // tile_dim0 = nelem
    g1[4] = 1;                                             // tile_dim1=1, dim2=0
    g1[5] = nelem;                                         // dim0_stride[31:0]
    g1[6] = 0;
    g1[7] = 0;

    const i32x4 z4 = {0, 0, 0, 0};
#if defined(__clang_major__) && (__clang_major__ >= 23)
    const i32x8 z8 = {0, 0, 0, 0, 0, 0, 0, 0};
    __builtin_amdgcn_tensor_load_to_lds(g0, g1, z4, z4, z8, 0);
#else
    __builtin_amdgcn_tensor_load_to_lds(g0, g1, z4, z4, 0);
#endif
}
#else
#define HAVE_TDM 0
#endif

// ---------------------------------------------------------------------------
// Shared 16xK x Kx64 WMMA tile core for the two projection GEMMs.
// Wave covers 64 output columns (4 tiles of 16), full K=512 in steps of 32.
// A fragment (16x32 bf16): lane = row (L&15); K groups per ISA layout:
//   lanes 0-15 : VGPR0-3 -> K=0..7, VGPR4-7 -> K=16..23
//   lanes16-31 : VGPR0-3 -> K=8..15, VGPR4-7 -> K=24..31
// B fragment (32x16 bf16): lane = column (L&15), 16 K-values in the same K groups as A, 2/VGPR.
// ---------------------------------------------------------------------------
__device__ __forceinline__ void proj_gemm(const float* __restrict__ x,
                                          const float* __restrict__ Bm,
                                          int row0, int waveCol, int lane,
                                          v8f acc[4])
{
    const int r = lane & 15, half = lane >> 4;
    const float* xr = x + (size_t)(row0 + r) * DIN;
    for (int k0 = 0; k0 < DIN; k0 += 32) {
        v16bf af;
        const float* xp = xr + k0 + half * 8;
#pragma unroll
        for (int i = 0; i < 8; ++i) af[i] = (__bf16)xp[i];
#pragma unroll
        for (int i = 0; i < 8; ++i) af[8 + i] = (__bf16)xp[16 + i];

        const float* bp = Bm + (size_t)(k0 + half * 8) * DOUT + waveCol + r;
#pragma unroll
        for (int t = 0; t < 4; ++t) {
            v16bf bf;
#pragma unroll
            for (int i = 0; i < 16; ++i) bf[i] = (__bf16)bp[(size_t)(i < 8 ? i : i + 8) * DOUT + t * 16];
            acc[t] = __builtin_amdgcn_wmma_f32_16x16x32_bf16(
                false, af, false, bf, (short)0, acc[t], false, false);
        }
    }
}

// h = lrelu(x @ W) -> bf16 (WMMA-B-ready row-major), wh1/wh2 fused dots.
__global__ void __launch_bounds__(128)
proj_h_kernel(const float* __restrict__ x, const float* __restrict__ W,
              const float* __restrict__ avec, __hip_bfloat16* __restrict__ hbf,
              float* __restrict__ wh1, float* __restrict__ wh2)
{
    __shared__ float s_w1[16], s_w2[16];
    const int tid = threadIdx.x, lane = tid & 31, wave = tid >> 5;
    const int r = lane & 15, half = lane >> 4;
    const int row0 = blockIdx.x * 16, waveCol = wave * 64;
    if (tid < 16) { s_w1[tid] = 0.f; s_w2[tid] = 0.f; }

    v8f acc[4];
#pragma unroll
    for (int t = 0; t < 4; ++t) acc[t] = v8f_zero();
    proj_gemm(x, W, row0, waveCol, lane, acc);
    __syncthreads();  // s_w* init visible before atomics

    float d1[8], d2[8];
#pragma unroll
    for (int v = 0; v < 8; ++v) { d1[v] = 0.f; d2[v] = 0.f; }
#pragma unroll
    for (int t = 0; t < 4; ++t) {
        const int col = waveCol + t * 16 + r;
        const float a1 = avec[col], a2 = avec[DOUT + col];
#pragma unroll
        for (int v = 0; v < 8; ++v) {
            const float hv = lrelu(acc[t][v]);                 // C/D: VGPR v = row
            const int row = row0 + v + 8 * half;
            hbf[(size_t)row * DOUT + col] = __float2bfloat16(hv);
            d1[v] += hv * a1;
            d2[v] += hv * a2;
        }
    }
#pragma unroll
    for (int v = 0; v < 8; ++v) {
        atomicAdd(&s_w1[v + 8 * half], d1[v]);
        atomicAdd(&s_w2[v + 8 * half], d2[v]);
    }
    __syncthreads();
    if (tid < 16) { wh1[row0 + tid] = s_w1[tid]; wh2[row0 + tid] = s_w2[tid]; }
}

// base = x @ res_w + res_b + bias  (fp32)
__global__ void __launch_bounds__(128)
proj_res_kernel(const float* __restrict__ x, const float* __restrict__ Rw,
                const float* __restrict__ resb, const float* __restrict__ bias,
                float* __restrict__ baseo)
{
    const int tid = threadIdx.x, lane = tid & 31, wave = tid >> 5;
    const int r = lane & 15, half = lane >> 4;
    const int row0 = blockIdx.x * 16, waveCol = wave * 64;
    v8f acc[4];
#pragma unroll
    for (int t = 0; t < 4; ++t) acc[t] = v8f_zero();
    proj_gemm(x, Rw, row0, waveCol, lane, acc);
#pragma unroll
    for (int t = 0; t < 4; ++t) {
        const int col = waveCol + t * 16 + r;
        const float add = resb[col] + bias[col];
#pragma unroll
        for (int v = 0; v < 8; ++v)
            baseo[(size_t)(row0 + v + 8 * half) * DOUT + col] = acc[t][v] + add;
    }
}

// M2 = max_j wh2[j]
__global__ void max_kernel(const float* __restrict__ wh2, float* __restrict__ M2)
{
    __shared__ float sm[256];
    float mx = -3.4e38f;
    for (int i = threadIdx.x; i < GN; i += 256) mx = fmaxf(mx, wh2[i]);
    sm[threadIdx.x] = mx;
    __syncthreads();
    for (int s = 128; s > 0; s >>= 1) {
        if (threadIdx.x < s) sm[threadIdx.x] = fmaxf(sm[threadIdx.x], sm[threadIdx.x + s]);
        __syncthreads();
    }
    if (threadIdx.x == 0) *M2 = sm[0];
}

// ---------------------------------------------------------------------------
// Fused masked-softmax + aggregation + L2-normalize + residual add.
// 16 rows / workgroup; 4 waves in a 2(K-split) x 2(N-split) grid:
//   kslot = wave&1  -> processes j-blocks j0 = kslot*32, stride 64
//   nslot = wave>>1 -> owns output columns [nslot*128, nslot*128+128)
// wh2 is DMA'd to LDS once per workgroup by the TDM (tensor_load_to_lds +
// s_wait_tensorcnt).  Per 32-neighbor block: build bf16 probability
// A-fragment in registers (p = exp(lrelu(wh1_i + wh2_j) - m_i), 0 if masked),
// 8 WMMAs vs L2-resident bf16 h.  Partial accs combined in LDS, then
// l-divide, L2-norm, + base.
// ---------------------------------------------------------------------------
__global__ void __launch_bounds__(128)
agg_kernel(const int* __restrict__ adj, const float* __restrict__ wh1,
           const float* __restrict__ wh2, const float* __restrict__ M2p,
           const __hip_bfloat16* __restrict__ hbf,
           const float* __restrict__ baseo, float* __restrict__ out)
{
    __shared__ float s_wh2[GN];         // 32 KB, TDM-staged
    __shared__ float s_acc[16][DOUT];   // 16 KB
    __shared__ float s_l[2][32];
    __shared__ float s_n2[16];
    const int tid = threadIdx.x, lane = tid & 31, wave = tid >> 5;
    const int r = lane & 15, half = lane >> 4;
    const int kslot = wave & 1, nslot = wave >> 1;
    const int row0 = blockIdx.x * 16;
    const int colBase = nslot * 128;

    for (int i = tid; i < 16 * DOUT; i += 128) (&s_acc[0][0])[i] = 0.f;
    if (tid < 16) s_n2[tid] = 0.f;

    // Stage wh2 (32 KB) into LDS: wave 0 drives the Tensor Data Mover.
#if HAVE_TDM
    if (wave == 0) {
        tdm_load_row_f32((unsigned)(unsigned long long)(const void*)s_wh2,
                         wh2, GN);
        __builtin_amdgcn_s_wait_tensorcnt(0);
    }
#else
    for (int i = tid; i < GN; i += 128) s_wh2[i] = wh2[i];
#endif
    __syncthreads();   // wh2 staged + s_acc/s_n2 init visible

    const float M2   = *M2p;
    const float wh1v = wh1[row0 + r];
    const float m    = lrelu(wh1v + M2);   // >= every masked score of this row

    v8f acc[8];
#pragma unroll
    for (int t = 0; t < 8; ++t) acc[t] = v8f_zero();
    float lpart = 0.f;
    const int kb0 = half * 8, kb1 = 16 + half * 8;   // A-frag K groups
    const size_t arow = (size_t)(row0 + r) * (size_t)GN;

    for (int j0 = kslot * 32; j0 < GN; j0 += 64) {
        const int* ap = adj + arow + j0;
        __builtin_prefetch((const void*)(ap + 64), 0, 1);   // next adj block
        int   av[16];
        float wv[16];
        *(int4*)(av + 0)    = *(const int4*)(ap + kb0);
        *(int4*)(av + 4)    = *(const int4*)(ap + kb0 + 4);
        *(int4*)(av + 8)    = *(const int4*)(ap + kb1);
        *(int4*)(av + 12)   = *(const int4*)(ap + kb1 + 4);
        *(float4*)(wv + 0)  = *(const float4*)(s_wh2 + j0 + kb0);
        *(float4*)(wv + 4)  = *(const float4*)(s_wh2 + j0 + kb0 + 4);
        *(float4*)(wv + 8)  = *(const float4*)(s_wh2 + j0 + kb1);
        *(float4*)(wv + 12) = *(const float4*)(s_wh2 + j0 + kb1 + 4);

        v16bf af;
#pragma unroll
        for (int i = 0; i < 16; ++i) {
            const float e = lrelu(wh1v + wv[i]);
            const float p = (av[i] > 0) ? __expf(e - m) : 0.f;
            lpart += p;
            af[i] = (__bf16)p;
        }

        const __bf16* hr = (const __bf16*)hbf + (size_t)(j0 + half * 8) * DOUT + colBase + r;
#pragma unroll
        for (int t = 0; t < 8; ++t) {
            v16bf bf;
#pragma unroll
            for (int i = 0; i < 16; ++i) bf[i] = hr[(size_t)(i < 8 ? i : i + 8) * DOUT + t * 16];
            acc[t] = __builtin_amdgcn_wmma_f32_16x16x32_bf16(
                false, af, false, bf, (short)0, acc[t], false, false);
        }
    }

    if (nslot == 0) s_l[kslot][lane] = lpart;   // l is identical across nslots
    __syncthreads();

    // Combine K-split partial accumulators in LDS.
#pragma unroll
    for (int t = 0; t < 8; ++t) {
        const int col = colBase + t * 16 + r;
#pragma unroll
        for (int v = 0; v < 8; ++v)
            atomicAdd(&s_acc[v + 8 * half][col], acc[t][v]);
    }
    __syncthreads();

    float hp[8][8];
    if (kslot == 0) {   // waves 0 & 2 own the finalize for their column half
        float invl[8], n2p[8];
#pragma unroll
        for (int v = 0; v < 8; ++v) {
            const int rr = v + 8 * half;
            const float lr = s_l[0][rr] + s_l[0][rr + 16] + s_l[1][rr] + s_l[1][rr + 16];
            invl[v] = 1.f / fmaxf(lr, 1e-30f);
            n2p[v]  = 0.f;
        }
#pragma unroll
        for (int t = 0; t < 8; ++t) {
            const int col = colBase + t * 16 + r;
#pragma unroll
            for (int v = 0; v < 8; ++v) {
                const float h = s_acc[v + 8 * half][col] * invl[v];
                hp[t][v] = h;
                n2p[v] += h * h;
            }
        }
#pragma unroll
        for (int v = 0; v < 8; ++v) atomicAdd(&s_n2[v + 8 * half], n2p[v]);
    }
    __syncthreads();
    if (kslot == 0) {
#pragma unroll
        for (int v = 0; v < 8; ++v) {
            const int row = row0 + v + 8 * half;
            const float innrm = 1.f / fmaxf(sqrtf(s_n2[v + 8 * half]), EPSN);
#pragma unroll
            for (int t = 0; t < 8; ++t) {
                const int col = colBase + t * 16 + r;
                out[(size_t)row * DOUT + col] =
                    hp[t][v] * innrm + baseo[(size_t)row * DOUT + col];
            }
        }
    }
}

extern "C" void kernel_launch(void* const* d_in, const int* in_sizes, int n_in,
                              void* d_out, int out_size, void* d_ws, size_t ws_size,
                              hipStream_t stream)
{
    (void)in_sizes; (void)n_in; (void)out_size; (void)ws_size;
    const float* x    = (const float*)d_in[0];
    const int*   adj  = (const int*)d_in[1];
    const float* W    = (const float*)d_in[2];
    const float* avec = (const float*)d_in[3];
    const float* bias = (const float*)d_in[4];
    const float* resw = (const float*)d_in[5];
    const float* resb = (const float*)d_in[6];
    float* out = (float*)d_out;

    // Workspace layout (~12.1 MB):
    char* ws = (char*)d_ws;
    __hip_bfloat16* hbf = (__hip_bfloat16*)ws;                    // 4 MB: h in bf16
    float* baseo = (float*)(ws + (size_t)GN * DOUT * 2);          // 8 MB: residual+biases
    float* wh1   = (float*)(ws + (size_t)GN * DOUT * 6);          // 32 KB
    float* wh2   = wh1 + GN;                                      // 32 KB
    float* M2    = wh2 + GN;                                      // 4 B

    const dim3 blk(128);
    proj_h_kernel  <<<GN / 16, blk, 0, stream>>>(x, W, avec, hbf, wh1, wh2);
    proj_res_kernel<<<GN / 16, blk, 0, stream>>>(x, resw, resb, bias, baseo);
    max_kernel     <<<1, 256, 0, stream>>>(wh2, M2);
    agg_kernel     <<<GN / 16, blk, 0, stream>>>(adj, wh1, wh2, M2, hbf, baseo, out);
}
